// SelfAttentionLayer_14568529068322
// MI455X (gfx1250) — compile-verified, matrix operand order checked by reading, not run on hardware
//
#include <hip/hip_runtime.h>

// ---------------- problem constants ----------------
#define Bn  2
#define Tn  2048
#define Hn  1024
#define NHn 16
#define HSn 64
#define MR  (Bn*Tn)          // 4096 rows for the projection GEMMs

typedef _Float16 f16;
typedef __attribute__((ext_vector_type(16))) _Float16 v16h;
typedef __attribute__((ext_vector_type(8)))  float    v8f;
typedef __attribute__((ext_vector_type(4)))  int      v4i;

union Frag { v16h v; unsigned u[8]; };

#ifndef __has_builtin
#define __has_builtin(x) 0
#endif
#if __has_builtin(__builtin_amdgcn_global_load_async_to_lds_b128) && \
    __has_builtin(__builtin_amdgcn_s_wait_asynccnt)
#define USE_ASYNC 1
#else
#define USE_ASYNC 0
#endif

#if USE_ASYNC
typedef __attribute__((address_space(1))) v4i g_v4i;   // global int4
typedef __attribute__((address_space(3))) v4i l_v4i;   // LDS int4
// one 16B per-lane async DMA: LDS[l] = MEM[g]   (tracked by ASYNCcnt)
__device__ __forceinline__ void async_b128(const void* g, void* l) {
  __builtin_amdgcn_global_load_async_to_lds_b128(
      (g_v4i*)(unsigned long long)g,
      (l_v4i*)(unsigned)(unsigned long long)l, 0, 0);
}
#define ASYNC_WAIT() __builtin_amdgcn_s_wait_asynccnt(0)
#else
#define ASYNC_WAIT() ((void)0)
#endif

// CDNA5 16-bit A/B fragment: VGPR v of lane-half h holds the packed k-pair
// starting at (v<4 ? 2v : 2v+8) + 8h   (ISA 7.12.2, 16-bit A 16x32 layout)
__device__ __forceinline__ int kpair(int v, int half) {
  return (v < 4 ? 2*v : 2*v + 8) + 8*half;
}

__device__ __forceinline__ unsigned pack2h(float a, float b) {
  union { f16 h[2]; unsigned u; } t;
  t.h[0] = (f16)a; t.h[1] = (f16)b;
  return t.u;
}

// ---- swizzled LDS layouts (16B-granule XOR swizzle: async-aligned + bank-clean)
// GEMM X tile: 128 rows x 32 halfs. dword index of element (r,k):
__device__ __forceinline__ int xdw(int r, int k) {
  return r*16 + ((((k >> 3) ^ ((r >> 2) & 3)) << 2) | ((k & 7) >> 1));
}
// Attention K tile: 64 rows x 64 halfs.
__device__ __forceinline__ int kdw(int r, int k) {
  return r*32 + ((((k >> 3) ^ ((r >> 1) & 7)) << 2) | ((k & 7) >> 1));
}

// ---------------- fp32 -> fp16 conversion ----------------
__global__ void convert_kernel(const float* __restrict__ in, f16* __restrict__ out, int n) {
  int i = blockIdx.x * blockDim.x + threadIdx.x;
  if (i < n) out[i] = (f16)in[i];
}

// ---------------- per-head V mean (exact fallback for fully-masked rows) ----
__global__ void vmean_kernel(const f16* __restrict__ Vh, float* __restrict__ vm) {
  int bh = blockIdx.x;           // 0..B*NH-1
  int d  = threadIdx.x;          // 0..63
  const f16* p = Vh + (long)bh * Tn * HSn + d;
  float s = 0.f;
  for (int t = 0; t < Tn; ++t) s += (float)p[t * HSn];
  vm[bh * HSn + d] = s * (1.0f / (float)Tn);
}

// stage one 128x32 X tile: thread owns (row r, 16-half segment seg)
__device__ __forceinline__ void stage_x(const f16* __restrict__ X, int blockRow,
                                        int k0, unsigned* __restrict__ buf,
                                        int r, int seg) {
  const f16* src = X + (blockRow + r) * Hn + k0 + seg * 16;
#if USE_ASYNC
  async_b128(src,     &buf[xdw(r, seg * 16)]);
  async_b128(src + 8, &buf[xdw(r, seg * 16 + 8)]);
#else
  union { uint4 q[2]; unsigned u[8]; } t;
  t.q[0] = ((const uint4*)src)[0];
  t.q[1] = ((const uint4*)src)[1];
#pragma unroll
  for (int j = 0; j < 8; ++j) buf[xdw(r, seg * 16 + 2 * j)] = t.u[j];
#endif
}

// ---------------- QKV projection: out = x @ W^T + b, into [B,NH,T,HS] -------
// block = 256 threads (8 waves); block tile 128x64, wave tile 32x32 (2x2 WMMA)
// async double-buffered X staging; W streams from L2 (2MB, fully resident).
__global__ __launch_bounds__(256) void qkv_proj_kernel(
    const f16* __restrict__ Xb,
    const f16* __restrict__ Wqb, const f16* __restrict__ Wkb, const f16* __restrict__ Wvb,
    const float* __restrict__ bq, const float* __restrict__ bk, const float* __restrict__ bv,
    f16* __restrict__ Qh, f16* __restrict__ Kh, f16* __restrict__ Vh)
{
  const f16* W; const float* bias; f16* Out;
  if (blockIdx.z == 0)      { W = Wqb; bias = bq; Out = Qh; }
  else if (blockIdx.z == 1) { W = Wkb; bias = bk; Out = Kh; }
  else                      { W = Wvb; bias = bv; Out = Vh; }

  __shared__ unsigned Xs[2][128 * 16];       // double-buffered, swizzled

  const int tid  = threadIdx.x;
  const int wave = tid >> 5, lane = tid & 31;
  const int half = lane >> 4, l16 = lane & 15;
  const int wm = wave & 3, wn = wave >> 2;
  const int blockRow = blockIdx.x * 128;
  const int waveRow  = blockRow + wm * 32;
  const int waveCol  = blockIdx.y * 64 + wn * 32;

  v8f acc[2][2] = {};
  const int r = tid >> 1, seg = tid & 1;

  stage_x(Xb, blockRow, 0, Xs[0], r, seg);
  ASYNC_WAIT();
  __syncthreads();

  int cur = 0;
  for (int k0 = 0; k0 < Hn; k0 += 32, cur ^= 1) {
    if (k0 + 32 < Hn) stage_x(Xb, blockRow, k0 + 32, Xs[cur ^ 1], r, seg);
    const unsigned* buf = Xs[cur];

    Frag a[2], bb[2];
#pragma unroll
    for (int mt = 0; mt < 2; ++mt) {
      const int row = wm * 32 + mt * 16 + l16;
#pragma unroll
      for (int v = 0; v < 8; ++v)
        a[mt].u[v] = buf[xdw(row, kpair(v, half))];
    }
#pragma unroll
    for (int nt = 0; nt < 2; ++nt) {
      const int col = waveCol + nt * 16 + l16;   // row of W = column of W^T
      const f16* wrow = W + col * Hn + k0;
#pragma unroll
      for (int v = 0; v < 8; ++v)
        bb[nt].u[v] = *(const unsigned*)(wrow + kpair(v, half));
    }
#pragma unroll
    for (int mt = 0; mt < 2; ++mt)
#pragma unroll
      for (int nt = 0; nt < 2; ++nt)
        acc[mt][nt] = __builtin_amdgcn_wmma_f32_16x16x32_f16(
            false, a[mt].v, false, bb[nt].v, (short)0, acc[mt][nt], false, false);

    ASYNC_WAIT();
    __syncthreads();
  }

  // scatter into head-major [B,NH,T,HS] with bias
#pragma unroll
  for (int nt = 0; nt < 2; ++nt) {
    const int col = waveCol + nt * 16 + l16;
    const float bv_ = bias[col];
    const int h = col >> 6, d = col & 63;
#pragma unroll
    for (int mt = 0; mt < 2; ++mt)
#pragma unroll
      for (int i = 0; i < 8; ++i) {
        const int row = waveRow + mt * 16 + i + 8 * half;
        const int b = row >> 11, t = row & (Tn - 1);
        Out[((b * NHn + h) * Tn + t) * HSn + d] = (f16)(acc[mt][nt][i] + bv_);
      }
  }
}

// ---------------- RoPE on Q and K (in place, fp32 math) ----------------
__global__ void rope_kernel(f16* __restrict__ Qh, f16* __restrict__ Kh,
                            const float* __restrict__ rope) {
  const int i = blockIdx.x * blockDim.x + threadIdx.x;
  const int total = Bn * NHn * Tn * (HSn / 2);
  if (i >= total) return;
  const int d2 = i & 31;
  const int t  = (i >> 5) & (Tn - 1);
  const int bh = i >> 16;
  const float c = rope[(t * 32 + d2) * 2 + 0];
  const float s = rope[(t * 32 + d2) * 2 + 1];
  const int base = (bh * Tn + t) * HSn + 2 * d2;
  float qe = (float)Qh[base], qo = (float)Qh[base + 1];
  Qh[base]     = (f16)(c * qe - s * qo);
  Qh[base + 1] = (f16)(s * qe + c * qo);
  float ke = (float)Kh[base], ko = (float)Kh[base + 1];
  Kh[base]     = (f16)(c * ke - s * ko);
  Kh[base + 1] = (f16)(s * ke + c * ko);
}

// ---------------- flash attention -------------------------------------------
// grid (B*NH, T/128); block 256 = 8 waves; wave owns 16 queries.
// Sᵀ = K·Qᵀ so softmax stats are in-lane; yᵀ = Vᵀ·Pᵀ with in-lane P repack.
// K tile staged by async DMA (overlaps the in-VGPR V transpose staging).
__global__ __launch_bounds__(256) void attn_kernel(
    const f16* __restrict__ Qh, const f16* __restrict__ Kh, const f16* __restrict__ Vh,
    const int* __restrict__ masks, const float* __restrict__ vmean,
    f16* __restrict__ Yb)
{
  __shared__ unsigned Ks32[64 * 32];   // K tile, swizzled (async destination)
  __shared__ f16 Vts[64 * 66];         // V^T [d][key], stride 66 (33 dwords, odd)
  __shared__ int Ms[64];
  unsigned* Vt32 = (unsigned*)Vts;

  const int tid  = threadIdx.x;
  const int wave = tid >> 5, lane = tid & 31;
  const int half = lane >> 4, l16 = lane & 15;
  const int bh = blockIdx.x;
  const int b  = bh >> 4;
  const int hh = bh & 15;
  const int qblk = blockIdx.y * 128;
  const int qw   = qblk + wave * 16;
  const int qg   = qw + l16;                 // this lane's query index

  const f16* Qp = Qh + (long)bh * Tn * HSn;
  const f16* Kp = Kh + (long)bh * Tn * HSn;
  const f16* Vp = Vh + (long)bh * Tn * HSn;
  const int* mp = masks + b * Tn;

  // Q held for the whole kernel as B-operand fragments (d split 0..31 / 32..63)
  Frag qb[2];
#pragma unroll
  for (int f = 0; f < 2; ++f)
#pragma unroll
    for (int v = 0; v < 8; ++v)
      qb[f].u[v] = *(const unsigned*)(Qp + qg * HSn + f * 32 + kpair(v, half));

  float m_run = -3.0e38f, l_run = 0.f;
  v8f yacc[4] = {};

  const int nChunk = (qblk >> 6) + 2;        // causal early exit, uniform per block
  const int cr = tid & 63, cp = tid >> 6;

  for (int c = 0; c < nChunk; ++c) {
    const int kbase = c * 64;
    { // ---- staging: K via async DMA, V transposed through VGPRs, masks ----
      const f16* ksrc = Kp + (kbase + cr) * HSn + cp * 16;
#if USE_ASYNC
      async_b128(ksrc,     &Ks32[kdw(cr, cp * 16)]);
      async_b128(ksrc + 8, &Ks32[kdw(cr, cp * 16 + 8)]);
#else
      union { uint4 q[2]; unsigned u[8]; } tk;
      tk.q[0] = ((const uint4*)ksrc)[0];
      tk.q[1] = ((const uint4*)ksrc)[1];
#pragma unroll
      for (int j = 0; j < 8; ++j) Ks32[kdw(cr, cp * 16 + 2 * j)] = tk.u[j];
#endif
      union { uint4 q[2]; f16 h[16]; } tv;
      const uint4* vsrc = (const uint4*)(Vp + (kbase + cr) * HSn + cp * 16);
      tv.q[0] = vsrc[0]; tv.q[1] = vsrc[1];
#pragma unroll
      for (int j = 0; j < 16; ++j)
        Vts[(cp * 16 + j) * 66 + cr] = tv.h[j];

      if (tid < 64) Ms[tid] = mp[kbase + tid];
      if (c + 1 < nChunk)                    // global_prefetch_b8 of next V chunk
        __builtin_prefetch(Vp + (kbase + 64 + cr) * HSn + cp * 16, 0, 0);
    }
    ASYNC_WAIT();
    __syncthreads();

    // ---- Sᵀ tiles: lane = query, (reg,half) = key ----
    float P[4][8];
#pragma unroll
    for (int s = 0; s < 4; ++s) {
      Frag ka[2];
      const int krow = s * 16 + l16;
#pragma unroll
      for (int f = 0; f < 2; ++f)
#pragma unroll
        for (int v = 0; v < 8; ++v)
          ka[f].u[v] = Ks32[kdw(krow, f * 32 + kpair(v, half))];
      v8f sacc = {};
      sacc = __builtin_amdgcn_wmma_f32_16x16x32_f16(false, ka[0].v, false, qb[0].v, (short)0, sacc, false, false);
      sacc = __builtin_amdgcn_wmma_f32_16x16x32_f16(false, ka[1].v, false, qb[1].v, (short)0, sacc, false, false);
#pragma unroll
      for (int i = 0; i < 8; ++i) {
        const int klocal = s * 16 + i + 8 * half;
        const int kg = kbase + klocal;
        float vv = sacc[i] * 0.125f;                       // 1/sqrt(64)
        if (kg > qg || Ms[klocal] == 1) vv = -1.0e10f;     // causal + key mask
        P[s][i] = vv;
      }
    }

    // ---- online softmax (per query = per lane; merge halves via shfl 16) ----
    float mc = -3.0e38f;
#pragma unroll
    for (int s = 0; s < 4; ++s)
#pragma unroll
      for (int i = 0; i < 8; ++i) mc = fmaxf(mc, P[s][i]);
    mc = fmaxf(mc, __shfl_xor(mc, 16, 32));
    const float mnew = fmaxf(m_run, mc);
    const float corr = __expf(m_run - mnew);
    float lc = 0.f;
#pragma unroll
    for (int s = 0; s < 4; ++s)
#pragma unroll
      for (int i = 0; i < 8; ++i) { float p = __expf(P[s][i] - mnew); P[s][i] = p; lc += p; }
    lc += __shfl_xor(lc, 16, 32);
    l_run = l_run * corr + lc;
    m_run = mnew;
#pragma unroll
    for (int dm = 0; dm < 4; ++dm)
#pragma unroll
      for (int i = 0; i < 8; ++i) yacc[dm][i] *= corr;

    // ---- repack Pᵀ (D layout) into B-operand fragments: fully in-lane ----
    Frag pb[2];
#pragma unroll
    for (int f = 0; f < 2; ++f)
#pragma unroll
      for (int v = 0; v < 8; ++v) {
        const int s  = 2 * f + (v >= 4 ? 1 : 0);
        const int i0 = (v < 4) ? 2 * v : 2 * (v - 4);
        pb[f].u[v] = pack2h(P[s][i0], P[s][i0 + 1]);
      }

    // ---- yᵀ += Vᵀ · Pᵀ ----
#pragma unroll
    for (int dm = 0; dm < 4; ++dm) {
      const int drow = dm * 16 + l16;
#pragma unroll
      for (int f = 0; f < 2; ++f) {
        Frag va;
#pragma unroll
        for (int v = 0; v < 8; ++v)
          va.u[v] = Vt32[drow * 33 + ((f * 32 + kpair(v, half)) >> 1)];
        yacc[dm] = __builtin_amdgcn_wmma_f32_16x16x32_f16(
            false, va.v, false, pb[f].v, (short)0, yacc[dm], false, false);
      }
    }
    __syncthreads();
  }

  // degenerate (all keys masked) row == reference's uniform softmax over all T
  const bool degen = (m_run <= -9.9e9f);
  const float invl = 1.0f / l_run;
#pragma unroll
  for (int dm = 0; dm < 4; ++dm)
#pragma unroll
    for (int i = 0; i < 8; ++i) {
      const int d = dm * 16 + i + 8 * half;
      const float val = degen ? vmean[bh * HSn + d] : yacc[dm][i] * invl;
      Yb[(b * Tn + qg) * Hn + hh * HSn + d] = (f16)val;
    }
}

// ---------------- output projection: out = y @ Wo^T (fp32 out) ----------------
__global__ __launch_bounds__(256) void out_proj_kernel(
    const f16* __restrict__ Yb, const f16* __restrict__ Wob, float* __restrict__ Out)
{
  __shared__ unsigned Xs[2][128 * 16];

  const int tid  = threadIdx.x;
  const int wave = tid >> 5, lane = tid & 31;
  const int half = lane >> 4, l16 = lane & 15;
  const int wm = wave & 3, wn = wave >> 2;
  const int blockRow = blockIdx.x * 128;
  const int waveRow  = blockRow + wm * 32;
  const int waveCol  = blockIdx.y * 64 + wn * 32;

  v8f acc[2][2] = {};
  const int r = tid >> 1, seg = tid & 1;

  stage_x(Yb, blockRow, 0, Xs[0], r, seg);
  ASYNC_WAIT();
  __syncthreads();

  int cur = 0;
  for (int k0 = 0; k0 < Hn; k0 += 32, cur ^= 1) {
    if (k0 + 32 < Hn) stage_x(Yb, blockRow, k0 + 32, Xs[cur ^ 1], r, seg);
    const unsigned* buf = Xs[cur];

    Frag a[2], bb[2];
#pragma unroll
    for (int mt = 0; mt < 2; ++mt) {
      const int row = wm * 32 + mt * 16 + l16;
#pragma unroll
      for (int v = 0; v < 8; ++v)
        a[mt].u[v] = buf[xdw(row, kpair(v, half))];
    }
#pragma unroll
    for (int nt = 0; nt < 2; ++nt) {
      const int col = waveCol + nt * 16 + l16;
      const f16* wrow = Wob + col * Hn + k0;
#pragma unroll
      for (int v = 0; v < 8; ++v)
        bb[nt].u[v] = *(const unsigned*)(wrow + kpair(v, half));
    }
#pragma unroll
    for (int mt = 0; mt < 2; ++mt)
#pragma unroll
      for (int nt = 0; nt < 2; ++nt)
        acc[mt][nt] = __builtin_amdgcn_wmma_f32_16x16x32_f16(
            false, a[mt].v, false, bb[nt].v, (short)0, acc[mt][nt], false, false);

    ASYNC_WAIT();
    __syncthreads();
  }

#pragma unroll
  for (int nt = 0; nt < 2; ++nt) {
    const int col = waveCol + nt * 16 + l16;
#pragma unroll
    for (int mt = 0; mt < 2; ++mt)
#pragma unroll
      for (int i = 0; i < 8; ++i) {
        const int row = waveRow + mt * 16 + i + 8 * half;
        Out[row * Hn + col] = acc[mt][nt][i];
      }
  }
}

// ---------------- launch ----------------
extern "C" void kernel_launch(void* const* d_in, const int* in_sizes, int n_in,
                              void* d_out, int out_size, void* d_ws, size_t ws_size,
                              hipStream_t stream) {
  (void)in_sizes; (void)n_in; (void)out_size; (void)ws_size;
  const float* x     = (const float*)d_in[0];
  const int*   masks = (const int*)  d_in[1];
  const float* Wq    = (const float*)d_in[2];
  const float* bq    = (const float*)d_in[3];
  const float* Wk    = (const float*)d_in[4];
  const float* bk    = (const float*)d_in[5];
  const float* Wv    = (const float*)d_in[6];
  const float* bv    = (const float*)d_in[7];
  const float* Wo    = (const float*)d_in[8];
  const float* rope  = (const float*)d_in[9];
  float* out = (float*)d_out;

  const size_t NX = (size_t)MR * Hn;            // 4M
  const size_t NW = (size_t)Hn * Hn;            // 1M
  const size_t NQ = (size_t)Bn * NHn * Tn * HSn;// 4M
  f16* ws  = (f16*)d_ws;
  f16* xb  = ws;
  f16* wqb = xb  + NX;
  f16* wkb = wqb + NW;
  f16* wvb = wkb + NW;
  f16* wob = wvb + NW;
  f16* Qb  = wob + NW;
  f16* Kb  = Qb  + NQ;
  f16* Vb  = Kb  + NQ;
  f16* Yb  = Vb  + NQ;
  float* vm = (float*)(Yb + NQ);                // B*NH*HS floats

  convert_kernel<<<(int)((NX + 255) / 256), 256, 0, stream>>>(x,  xb,  (int)NX);
  convert_kernel<<<(int)((NW + 255) / 256), 256, 0, stream>>>(Wq, wqb, (int)NW);
  convert_kernel<<<(int)((NW + 255) / 256), 256, 0, stream>>>(Wk, wkb, (int)NW);
  convert_kernel<<<(int)((NW + 255) / 256), 256, 0, stream>>>(Wv, wvb, (int)NW);
  convert_kernel<<<(int)((NW + 255) / 256), 256, 0, stream>>>(Wo, wob, (int)NW);

  qkv_proj_kernel<<<dim3(MR / 128, Hn / 64, 3), 256, 0, stream>>>(
      xb, wqb, wkb, wvb, bq, bk, bv, Qb, Kb, Vb);

  const int npair = Bn * NHn * Tn * (HSn / 2);
  rope_kernel<<<(npair + 255) / 256, 256, 0, stream>>>(Qb, Kb, rope);

  vmean_kernel<<<Bn * NHn, HSn, 0, stream>>>(Vb, vm);

  attn_kernel<<<dim3(Bn * NHn, Tn / 128), 256, 0, stream>>>(Qb, Kb, Vb, masks, vm, Yb);

  out_proj_kernel<<<dim3(MR / 128, Hn / 64), 256, 0, stream>>>(Yb, wob, out);
}
